// SkilledLoRALinear_13365938225193
// MI455X (gfx1250) — compile-verified, matrix operand order checked by reading, not run on hardware
//
#include <hip/hip_runtime.h>
#include <hip/hip_bf16.h>

// ---- problem constants (from reference) ----
#define B_    16
#define S_    4096
#define INF_  1024
#define OUTF_ 1024
#define NSK_  16
#define R_    16

// ---- own vector types (trivially copyable, bit_cast-safe) ----
typedef __bf16 bf16;
typedef __attribute__((ext_vector_type(16))) __bf16        v16bf;
typedef __attribute__((ext_vector_type(8)))  float         v8f;
typedef __attribute__((ext_vector_type(8)))  unsigned int  v8u;
typedef __attribute__((ext_vector_type(4)))  unsigned int  u32x4;
typedef __attribute__((ext_vector_type(4)))  float         f32x4;

__device__ __forceinline__ v16bf mkfrag(u32x4 lo, u32x4 hi) {
  v8u u;
  u[0]=lo.x; u[1]=lo.y; u[2]=lo.z; u[3]=lo.w;
  u[4]=hi.x; u[5]=hi.y; u[6]=hi.z; u[7]=hi.w;
  return __builtin_bit_cast(v16bf, u);
}

__device__ __forceinline__ unsigned pk2(float a, float b) {
  bf16 x = (bf16)a, y = (bf16)b;
  unsigned short ux = __builtin_bit_cast(unsigned short, x);
  unsigned short uy = __builtin_bit_cast(unsigned short, y);
  return (unsigned)ux | ((unsigned)uy << 16);
}

// -------- prep 1: routing (sigmoid + normalize) --------
__global__ void k_route(const float* __restrict__ skill_logits,
                        const int*   __restrict__ task_ids,
                        float*       __restrict__ logits) {
  int b = threadIdx.x;
  if (b < B_) {
    int t = task_ids[b];
    float v[NSK_]; float s = 0.f;
    #pragma unroll
    for (int k = 0; k < NSK_; ++k) {
      float x  = skill_logits[t * NSK_ + k];
      float sg = 1.f / (1.f + __expf(-x));
      v[k] = sg; s += sg;
    }
    float inv = 1.f / (s + 1e-12f);
    #pragma unroll
    for (int k = 0; k < NSK_; ++k) logits[b * NSK_ + k] = v[k] * inv;
  }
}

// -------- prep 2: weight fp32 -> bf16, WMMA-fragment layout --------
// wpack[(kt*1024 + o)*32 + kk] = bf16(weight[o][kt*32+kk])   (column-of-32-K contiguous)
__global__ void k_packW(const float* __restrict__ w, bf16* __restrict__ wpack) {
  int idx = blockIdx.x * 256 + threadIdx.x;          // 1M elements
  int kk = idx & 31;
  int o  = (idx >> 5) & 1023;
  int kt = idx >> 15;
  wpack[idx] = (bf16)w[o * 1024 + kt * 32 + kk];
}

// -------- prep 3: per-batch LoRA mixes --------
// wApack[((b*32+kt)*16 + r)*32 + kk] = bf16( sum_k logits[b,k] * A[k, (kt*32+kk)*16 + r] )
// wBg[b*16384 + r*1024 + o]          =       sum_k logits[b,k] * Bsk[k, r*1024 + o]
__global__ void k_packAB(const float* __restrict__ logits,
                         const float* __restrict__ A,
                         const float* __restrict__ Bsk,
                         bf16*  __restrict__ wApack,
                         float* __restrict__ wBg) {
  int idx = blockIdx.x * 256 + threadIdx.x;          // 512K threads total
  if (idx < 262144) {                                // wApack
    int kk = idx & 31, r = (idx >> 5) & 15, kt = (idx >> 9) & 31, b = idx >> 14;
    int i = kt * 32 + kk;
    const float* lg = logits + b * NSK_;
    float acc = 0.f;
    #pragma unroll
    for (int k = 0; k < NSK_; ++k) acc += lg[k] * A[k * (R_ * INF_) + i * R_ + r];
    wApack[idx] = (bf16)acc;
  } else {                                           // wBg
    int j = idx - 262144;
    int o = j & 1023, r = (j >> 10) & 15, b = j >> 14;
    const float* lg = logits + b * NSK_;
    float acc = 0.f;
    #pragma unroll
    for (int k = 0; k < NSK_; ++k) acc += lg[k] * Bsk[k * (R_ * OUTF_) + r * OUTF_ + o];
    wBg[j] = acc;
  }
}

// -------- main fused kernel --------
// 128x128 output tile per 256-thread block (8 wave32). Per K-step(32):
//   stage A (fp32->bf16) into LDS; each wave: 8 base WMMAs + 2 LoRA-t WMMAs.
// Epilogue: t -> LDS, then out = base + bias + (1/16) * t . wB  (16 FMAs/elem).
__global__ __launch_bounds__(256)
void k_main(const float* __restrict__ inp, const float* __restrict__ bias,
            const bf16* __restrict__ wpack, const bf16* __restrict__ wApack,
            const float* __restrict__ wBg, float* __restrict__ out) {
  __shared__ __align__(16) unsigned char ldsA[128 * 80];   // bf16 A tile, padded stride
  __shared__ __align__(16) float t_lds[128 * 16];          // t = input . wA
  __shared__ __align__(16) float wB_lds[16 * 128];         // wB columns for this tile
  __shared__ float bias_lds[128];

  const int bid   = blockIdx.x;
  const int nTile = bid & 7;
  const int sTile = (bid >> 3) & 31;
  const int b     = bid >> 8;
  const int sBase = sTile * 128;
  const int oBase = nTile * 128;

  const int tid   = threadIdx.x;
  const int lane  = tid & 31;
  const int w     = tid >> 5;
  const int waveM = w & 3;
  const int waveN = w >> 2;
  const int lrow  = lane & 15;
  const int lhalf = lane >> 4;

  // stage wB tile + bias once (covered by first loop barrier)
  for (int e = tid; e < 16 * 128; e += 256)
    wB_lds[e] = wBg[(size_t)b * 16384 + (size_t)(e >> 7) * 1024 + oBase + (e & 127)];
  if (tid < 128) bias_lds[tid] = bias[oBase + tid];

  v8f zacc = {};
  v8f accB[2][4];
  v8f accT[2];
  #pragma unroll
  for (int mi = 0; mi < 2; ++mi) {
    accT[mi] = zacc;
    #pragma unroll
    for (int ni = 0; ni < 4; ++ni) accB[mi][ni] = zacc;
  }

  const int arow = tid >> 1, ahalf = tid & 1;
  const float* asrc = inp + ((size_t)(b * S_ + sBase + arow)) * INF_ + ahalf * 16;

  for (int kt = 0; kt < 32; ++kt) {
    // ---- stage A tile: 128 rows x 32 K, fp32 -> bf16 ----
    {
      const f32x4* p = (const f32x4*)(asrc + kt * 32);
      f32x4 f0 = p[0], f1 = p[1], f2 = p[2], f3 = p[3];
      u32x4 u0, u1;
      u0.x = pk2(f0.x, f0.y); u0.y = pk2(f0.z, f0.w);
      u0.z = pk2(f1.x, f1.y); u0.w = pk2(f1.z, f1.w);
      u1.x = pk2(f2.x, f2.y); u1.y = pk2(f2.z, f2.w);
      u1.z = pk2(f3.x, f3.y); u1.w = pk2(f3.z, f3.w);
      u32x4* d = (u32x4*)(ldsA + arow * 80 + ahalf * 32);
      d[0] = u0; d[1] = u1;
      if (kt < 31) __builtin_prefetch(asrc + (kt + 1) * 32, 0, 3);  // global_prefetch_b8
    }
    __syncthreads();

    // ---- A fragments from LDS (ISA 16-bit A layout: lane<16 K0-7/16-23, lane>=16 K8-15/24-31)
    v16bf aF[2];
    #pragma unroll
    for (int mi = 0; mi < 2; ++mi) {
      const unsigned char* p = ldsA + (waveM * 32 + mi * 16 + lrow) * 80 + lhalf * 16;
      aF[mi] = mkfrag(*(const u32x4*)p, *(const u32x4*)(p + 32));
    }
    // ---- B fragments straight from L2-resident packed weight
    v16bf bF[4];
    #pragma unroll
    for (int ni = 0; ni < 4; ++ni) {
      int co = oBase + waveN * 64 + ni * 16 + lrow;
      const unsigned char* q = (const unsigned char*)wpack + ((size_t)(kt * 1024 + co)) * 64 + lhalf * 16;
      bF[ni] = mkfrag(*(const u32x4*)q, *(const u32x4*)(q + 32));
    }
    // ---- wA fragment (16 LoRA columns)
    v16bf wF;
    {
      const unsigned char* q = (const unsigned char*)wApack + ((size_t)((b * 32 + kt) * 16 + lrow)) * 64 + lhalf * 16;
      wF = mkfrag(*(const u32x4*)q, *(const u32x4*)(q + 32));
    }

    #pragma unroll
    for (int mi = 0; mi < 2; ++mi)
      #pragma unroll
      for (int ni = 0; ni < 4; ++ni)
        accB[mi][ni] = __builtin_amdgcn_wmma_f32_16x16x32_bf16(
            false, aF[mi], false, bF[ni], (short)0, accB[mi][ni], false, false);
    #pragma unroll
    for (int mi = 0; mi < 2; ++mi)
      accT[mi] = __builtin_amdgcn_wmma_f32_16x16x32_bf16(
          false, aF[mi], false, wF, (short)0, accT[mi], false, false);

    __syncthreads();
  }

  // ---- share t across waves (waveN duplicates -> only waveN==0 writes)
  if (waveN == 0) {
    #pragma unroll
    for (int mi = 0; mi < 2; ++mi)
      #pragma unroll
      for (int v = 0; v < 8; ++v)
        t_lds[(waveM * 32 + mi * 16 + v + lhalf * 8) * 16 + lrow] = accT[mi][v];
  }
  __syncthreads();

  // ---- epilogue: out = base + bias + (1/R) * t . wB
  const float scaling = 1.0f / (float)R_;
  #pragma unroll
  for (int ni = 0; ni < 4; ++ni) {
    int nl = waveN * 64 + ni * 16 + lrow;          // local column
    float wbc[16];
    #pragma unroll
    for (int r = 0; r < 16; ++r) wbc[r] = wB_lds[r * 128 + nl];
    float bv = bias_lds[nl];
    #pragma unroll
    for (int mi = 0; mi < 2; ++mi) {
      #pragma unroll
      for (int v = 0; v < 8; ++v) {
        int rl = waveM * 32 + mi * 16 + v + lhalf * 8;   // local row
        const f32x4* tp = (const f32x4*)&t_lds[rl * 16];
        f32x4 t0 = tp[0], t1 = tp[1], t2 = tp[2], t3 = tp[3];
        float s = t0.x*wbc[0] + t0.y*wbc[1] + t0.z*wbc[2]  + t0.w*wbc[3]
                + t1.x*wbc[4] + t1.y*wbc[5] + t1.z*wbc[6]  + t1.w*wbc[7]
                + t2.x*wbc[8] + t2.y*wbc[9] + t2.z*wbc[10] + t2.w*wbc[11]
                + t3.x*wbc[12]+ t3.y*wbc[13]+ t3.z*wbc[14] + t3.w*wbc[15];
        out[((size_t)(b * S_ + sBase + rl)) * OUTF_ + oBase + nl]
            = accB[mi][ni][v] + bv + s * scaling;
      }
    }
  }
}

// -------- workspace layout (needs ~3.7 MB) --------
#define WS_LOGITS  0u
#define WS_WB      4096u                        // 16*16*1024 f32 = 1 MB
#define WS_WAPACK  (4096u + 1048576u)           // 16*32*16*32 bf16 = 512 KB
#define WS_WPACK   (4096u + 1048576u + 524288u) // 1024*1024 bf16 = 2 MB

extern "C" void kernel_launch(void* const* d_in, const int* in_sizes, int n_in,
                              void* d_out, int out_size, void* d_ws, size_t ws_size,
                              hipStream_t stream) {
  const float* input        = (const float*)d_in[0];
  const float* skill_logits = (const float*)d_in[1];
  const float* weight       = (const float*)d_in[2];
  const float* bias         = (const float*)d_in[3];
  const float* skills_A     = (const float*)d_in[4];
  const float* skills_B     = (const float*)d_in[5];
  const int*   task_ids     = (const int*)d_in[6];
  float*       out          = (float*)d_out;

  char* ws = (char*)d_ws;
  float* logits = (float*)(ws + WS_LOGITS);
  float* wBg    = (float*)(ws + WS_WB);
  bf16*  wApack = (bf16*)(ws + WS_WAPACK);
  bf16*  wpack  = (bf16*)(ws + WS_WPACK);

  k_route <<<1,    32,  0, stream>>>(skill_logits, task_ids, logits);
  k_packW <<<4096, 256, 0, stream>>>(weight, wpack);
  k_packAB<<<2048, 256, 0, stream>>>(logits, skills_A, skills_B, wApack, wBg);
  // grid: 16 batches * 32 S-tiles * 8 N-tiles; N-tiles adjacent for L2 input reuse
  k_main  <<<4096, 256, 0, stream>>>(input, bias, wpack, wApack, wBg, out);
}
